// Qwen3MoeAttention_3109556322627
// MI455X (gfx1250) — matrix operand order repaired and checked by reading, not compiled, not run on hardware
//
#include <hip/hip_runtime.h>
#include <hip/hip_bf16.h>
#include <stdint.h>

#define B_   2
#define S_   2048
#define H_   4096
#define NH_  32
#define NKV_ 8
#define HD_  128

typedef __attribute__((ext_vector_type(16))) __bf16      v16bf;
typedef __attribute__((ext_vector_type(8)))  float        v8f;
typedef __attribute__((ext_vector_type(4)))  unsigned int u32x4;

union Frag { v16bf v; u32x4 q[2]; };   // 32 bytes, trivial members

__device__ __forceinline__ unsigned short f2bf(float f) {
  unsigned u = __builtin_bit_cast(unsigned, f);
  u += 0x7fffu + ((u >> 16) & 1u);     // round-to-nearest-even
  return (unsigned short)(u >> 16);
}

__device__ __forceinline__ v8f wmma_bf16(v16bf a, v16bf b, v8f c) {
  return __builtin_amdgcn_wmma_f32_16x16x32_bf16(
      /*neg_a=*/false, a, /*neg_b=*/false, b,
      /*c_mod=*/(short)0, c, /*reuse_a=*/false, /*reuse_b=*/false);
}

// ---------------------------------------------------------------- convert
__global__ void k_f32_to_bf16(const float* __restrict__ in,
                              unsigned short* __restrict__ out, int n) {
  int i = blockIdx.x * blockDim.x + threadIdx.x;
  if (i < n) out[i] = f2bf(in[i]);
}

// ---------------------------------------------------------------- GEMM
// C(M,N) f32 = A(M,K) bf16 row-major  x  W(N,K)^T bf16 row-major.
// One wave computes a 64x64 C tile (16 WMMAs per K-step of 32).
// K-loop is branch-free so the compiler can unroll x2 and software-pipeline
// loads against the WMMA pipe (partial s_wait_loadcnt).
__global__ __launch_bounds__(256) void k_gemm_bf16(
    const unsigned short* __restrict__ A,
    const unsigned short* __restrict__ W,
    float* __restrict__ C, int M, int N, int K)
{
  const int lane = threadIdx.x & 31;
  const int half = lane >> 4, nl = lane & 15;
  const int wavesPerBlock = blockDim.x >> 5;
  const int wid = blockIdx.x * wavesPerBlock + (threadIdx.x >> 5);
  const int ntn = N >> 6, ntm = M >> 6;
  const int ntot = ntm * ntn;
  const int stride = gridDim.x * wavesPerBlock;

  for (int t = wid; t < ntot; t += stride) {
    const int row0 = (t / ntn) << 6;
    const int col0 = (t % ntn) << 6;

    // one-time tile prefetch into GL2 (global_prefetch_b8), outside hot loop
#pragma unroll
    for (int i = 0; i < 4; ++i)
      __builtin_prefetch(A + (size_t)(row0 + 16 * i + nl) * K, 0, 3);
#pragma unroll
    for (int j = 0; j < 4; ++j)
      __builtin_prefetch(W + (size_t)(col0 + 16 * j + nl) * K, 0, 3);

    v8f acc[4][4];
    for (int i = 0; i < 4; ++i)
      for (int j = 0; j < 4; ++j)
        for (int r = 0; r < 8; ++r) acc[i][j][r] = 0.f;

#pragma unroll 2
    for (int k0 = 0; k0 < K; k0 += 32) {
      Frag a[4], b[4];
#pragma unroll
      for (int i = 0; i < 4; ++i) {
        const unsigned short* pa =
            A + (size_t)(row0 + 16 * i + nl) * K + k0 + 16 * half;
        a[i].q[0] = *(const u32x4*)pa;          // K: k0+16h .. +7
        a[i].q[1] = *(const u32x4*)(pa + 8);    // K: k0+16h+8 .. +15
      }
#pragma unroll
      for (int j = 0; j < 4; ++j) {
        const unsigned short* pb =
            W + (size_t)(col0 + 16 * j + nl) * K + k0 + 16 * half;
        b[j].q[0] = ((const u32x4*)pb)[0];      // K: k0+16h .. +15
        b[j].q[1] = ((const u32x4*)pb)[1];
      }
#pragma unroll
      for (int i = 0; i < 4; ++i)
#pragma unroll
        for (int j = 0; j < 4; ++j)
          acc[i][j] = wmma_bf16(a[i].v, b[j].v, acc[i][j]);
    }
#pragma unroll
    for (int i = 0; i < 4; ++i)
      for (int j = 0; j < 4; ++j)
        for (int r = 0; r < 8; ++r)
          C[(size_t)(row0 + 16 * i + 8 * half + r) * N + col0 + 16 * j + nl] =
              acc[i][j][r];
  }
}

// ------------------------------------------------- RMSNorm + RoPE (per head)
// One wave per (b, s, head): 128 elements, 4 per lane; RoPE partner d±64 is
// within the same lane (slot j±2); head-dim reduction via shfl_xor.
__global__ __launch_bounds__(256) void k_normrope(
    const float* __restrict__ in,      // (B*S, nh*HD)
    const float* __restrict__ nw,      // (HD)
    const float* __restrict__ cosb,    // (B,S,HD)
    const float* __restrict__ sinb,    // (B,S,HD)
    unsigned short* __restrict__ out,  // (B, nh, S, HD) bf16
    int nh)
{
  const int lane = threadIdx.x & 31;
  const int wave = blockIdx.x * (blockDim.x >> 5) + (threadIdx.x >> 5);
  const int h = wave % nh;
  const int s = (wave / nh) % S_;
  const int b = wave / (nh * S_);
  if (b >= B_) return;

  const float* row = in + (size_t)(b * S_ + s) * (nh * HD_) + h * HD_;
  float x[4], ss = 0.f;
#pragma unroll
  for (int j = 0; j < 4; ++j) { x[j] = row[lane + 32 * j]; ss += x[j] * x[j]; }
  for (int m = 16; m >= 1; m >>= 1) ss += __shfl_xor(ss, m, 32);
  const float r = rsqrtf(ss * (1.f / HD_) + 1e-6f);

  const float* cp = cosb + (size_t)(b * S_ + s) * HD_;
  const float* sp = sinb + (size_t)(b * S_ + s) * HD_;
  float xn[4];
#pragma unroll
  for (int j = 0; j < 4; ++j) xn[j] = x[j] * r * nw[lane + 32 * j];

  unsigned short* op = out + ((size_t)(b * nh + h) * S_ + s) * HD_;
#pragma unroll
  for (int j = 0; j < 4; ++j) {
    int d = lane + 32 * j;
    float rot = (j < 2) ? -xn[j + 2] : xn[j - 2];   // rotate_half pairing d<->d±64
    op[d] = f2bf(xn[j] * cp[d] + rot * sp[d]);
  }
}

// ------------------------------------------------- V: convert + transpose
// (B*S, NKV*HD) f32 -> (B, NKV, HD, S) bf16 via 32x32 LDS tile:
// coalesced reads along d, coalesced writes along s.
__global__ __launch_bounds__(256) void k_vtrans(
    const float* __restrict__ in,
    unsigned short* __restrict__ out)
{
  __shared__ float tile[32][33];            // +1 pad: no bank conflicts
  const int tx = threadIdx.x & 31;
  const int ty = threadIdx.x >> 5;          // 0..7
  int t = blockIdx.x;
  const int s0 = (t & 63) << 5;  t >>= 6;   // S/32  = 64
  const int d0 = (t & 3)  << 5;  t >>= 2;   // HD/32 = 4
  const int h  = t & 7;          t >>= 3;   // NKV
  const int b  = t;
#pragma unroll
  for (int i = 0; i < 4; ++i) {
    int s = s0 + ty + 8 * i;
    tile[ty + 8 * i][tx] =
        in[(size_t)(b * S_ + s) * (NKV_ * HD_) + h * HD_ + d0 + tx];
  }
  __syncthreads();
#pragma unroll
  for (int i = 0; i < 4; ++i) {
    int d = d0 + ty + 8 * i;
    out[((size_t)(b * NKV_ + h) * HD_ + d) * S_ + s0 + tx] =
        f2bf(tile[tx][ty + 8 * i]);
  }
}

// ------------------------------------------------- causal flash attention
// One wave handles 16 query rows; KV chunks of 64; online softmax.
// QK^T: Q A-frags + K "W-layout" B-frags.  PV: P staged through LDS
// (C-layout -> A-layout transpose), V^T gives K-contiguous B-frags.
__global__ __launch_bounds__(256) void k_flash(
    const unsigned short* __restrict__ Q,   // (B,NH,S,HD)
    const unsigned short* __restrict__ Kc,  // (B,NKV,S,HD)
    const unsigned short* __restrict__ Vt,  // (B,NKV,HD,S)
    unsigned short* __restrict__ O)         // (B*S, NH*HD)
{
  __shared__ __align__(16) unsigned short plds[8][16 * 64];
  const int wiB  = threadIdx.x >> 5;
  const int lane = threadIdx.x & 31;
  const int half = lane >> 4, nl = lane & 15;
  const int wave = blockIdx.x * 8 + wiB;
  const int QT = S_ / 16;
  const int qt = wave % QT;
  const int h  = (wave / QT) % NH_;
  const int b  =  wave / (QT * NH_);
  const int hkv = h >> 2;                 // NH/NKV = 4
  const int q0 = qt * 16;

  const unsigned short* Qb = Q + ((size_t)(b * NH_ + h) * S_ + q0) * HD_;
  const unsigned short* Kb = Kc + (size_t)(b * NKV_ + hkv) * S_ * HD_;
  const unsigned short* Vb = Vt + (size_t)(b * NKV_ + hkv) * HD_ * S_;

  Frag qf[4];
#pragma unroll
  for (int kd = 0; kd < 4; ++kd) {
    const unsigned short* p = Qb + (size_t)nl * HD_ + 32 * kd + 16 * half;
    qf[kd].q[0] = *(const u32x4*)p;
    qf[kd].q[1] = *(const u32x4*)(p + 8);
  }

  v8f o[8];
  float mr[8], lr[8];
  for (int t = 0; t < 8; ++t)
    for (int r = 0; r < 8; ++r) o[t][r] = 0.f;
  for (int r = 0; r < 8; ++r) { mr[r] = -3.0e38f; lr[r] = 0.f; }

  unsigned short* pl = &plds[wiB][0];
  const float scale = 0.08838834764831845f;   // 1/sqrt(128)

  for (int kv0 = 0; kv0 < q0 + 16; kv0 += 64) {
    // prefetch next KV chunk while computing this one (once per chunk)
    if (kv0 + 64 < q0 + 16) {
      __builtin_prefetch(Kb + (size_t)(kv0 + 64 + nl) * HD_ + half * 64, 0, 3);
      __builtin_prefetch(Vb + (size_t)(lane * 4) * S_ + kv0 + 64, 0, 3);
    }
    // ---- scores S = Q K^T (16x64): 16 WMMAs
    v8f sa[4];
#pragma unroll
    for (int nt = 0; nt < 4; ++nt) {
      for (int r = 0; r < 8; ++r) sa[nt][r] = 0.f;
#pragma unroll
      for (int kd = 0; kd < 4; ++kd) {
        Frag kf;
        const unsigned short* p =
            Kb + (size_t)(kv0 + 16 * nt + nl) * HD_ + 32 * kd + 16 * half;
        kf.q[0] = ((const u32x4*)p)[0];
        kf.q[1] = ((const u32x4*)p)[1];
        sa[nt] = wmma_bf16(qf[kd].v, kf.v, sa[nt]);
      }
    }
    // ---- scale + causal mask (only diagonal chunk needs it)
    const bool diag = (kv0 + 64 > q0);
#pragma unroll
    for (int nt = 0; nt < 4; ++nt)
      for (int r = 0; r < 8; ++r) {
        float sv = sa[nt][r] * scale;
        if (diag) {
          int col = kv0 + 16 * nt + nl;
          int row = q0 + 8 * half + r;
          sv = (col <= row) ? sv : -3.0e38f;
        }
        sa[nt][r] = sv;
      }
    // ---- online softmax row stats (row lives across 16 lanes of a half)
    float mnew[8], scl[8];
#pragma unroll
    for (int r = 0; r < 8; ++r) {
      float t = fmaxf(fmaxf(sa[0][r], sa[1][r]), fmaxf(sa[2][r], sa[3][r]));
      t = fmaxf(t, __shfl_xor(t, 1, 32));
      t = fmaxf(t, __shfl_xor(t, 2, 32));
      t = fmaxf(t, __shfl_xor(t, 4, 32));
      t = fmaxf(t, __shfl_xor(t, 8, 32));
      mnew[r] = fmaxf(mr[r], t);
      scl[r]  = __expf(mr[r] - mnew[r]);
      mr[r]   = mnew[r];
    }
    float ps[8];
    for (int r = 0; r < 8; ++r) ps[r] = 0.f;
#pragma unroll
    for (int nt = 0; nt < 4; ++nt)
      for (int r = 0; r < 8; ++r) {
        float p = __expf(sa[nt][r] - mnew[r]);
        ps[r] += p;
        pl[(8 * half + r) * 64 + 16 * nt + nl] = f2bf(p);  // stage P in LDS
      }
#pragma unroll
    for (int r = 0; r < 8; ++r) {
      float t = ps[r];
      t += __shfl_xor(t, 1, 32);
      t += __shfl_xor(t, 2, 32);
      t += __shfl_xor(t, 4, 32);
      t += __shfl_xor(t, 8, 32);
      lr[r] = lr[r] * scl[r] + t;
    }
#pragma unroll
    for (int t8 = 0; t8 < 8; ++t8)
      for (int r = 0; r < 8; ++r) o[t8][r] *= scl[r];

    asm volatile("s_wait_dscnt 0" ::: "memory");  // P stores visible before reads

    // ---- O += P V  (16x128): 16 WMMAs
#pragma unroll
    for (int kp = 0; kp < 2; ++kp) {
      Frag af;
      const unsigned short* ap = pl + (size_t)nl * 64 + 32 * kp + 16 * half;
      af.q[0] = *(const u32x4*)ap;
      af.q[1] = *(const u32x4*)(ap + 8);
#pragma unroll
      for (int dt = 0; dt < 8; ++dt) {
        Frag vf;
        const unsigned short* vp =
            Vb + (size_t)(16 * dt + nl) * S_ + kv0 + 32 * kp + 16 * half;
        vf.q[0] = ((const u32x4*)vp)[0];
        vf.q[1] = ((const u32x4*)vp)[1];
        o[dt] = wmma_bf16(af.v, vf.v, o[dt]);
      }
    }
  }

  unsigned short* ob = O + (size_t)(b * S_ + q0) * (NH_ * HD_) + h * HD_;
#pragma unroll
  for (int dt = 0; dt < 8; ++dt)
    for (int r = 0; r < 8; ++r)
      ob[(size_t)(8 * half + r) * (NH_ * HD_) + 16 * dt + nl] =
          f2bf(o[dt][r] / lr[r]);
}

// ----------------------------------------------------------------- launch
extern "C" void kernel_launch(void* const* d_in, const int* in_sizes, int n_in,
                              void* d_out, int out_size, void* d_ws, size_t ws_size,
                              hipStream_t stream) {
  (void)in_sizes; (void)n_in; (void)out_size; (void)ws_size;
  const float* x    = (const float*)d_in[0];
  const float* cosb = (const float*)d_in[1];
  const float* sinb = (const float*)d_in[2];
  const float* wq   = (const float*)d_in[3];
  const float* wk   = (const float*)d_in[4];
  const float* wv   = (const float*)d_in[5];
  const float* wo   = (const float*)d_in[6];
  const float* qnw  = (const float*)d_in[7];
  const float* knw  = (const float*)d_in[8];
  float* out = (float*)d_out;

  char* ws = (char*)d_ws;
  size_t off = 0;
  auto alloc = [&](size_t bytes) {
    char* p = ws + off;
    off += (bytes + 255) & ~(size_t)255;
    return p;
  };
  unsigned short* x_bf  = (unsigned short*)alloc((size_t)4096 * 4096 * 2);
  unsigned short* wq_bf = (unsigned short*)alloc((size_t)4096 * 4096 * 2);
  unsigned short* wk_bf = (unsigned short*)alloc((size_t)1024 * 4096 * 2);
  unsigned short* wv_bf = (unsigned short*)alloc((size_t)1024 * 4096 * 2);
  unsigned short* wo_bf = (unsigned short*)alloc((size_t)4096 * 4096 * 2);
  float*          qf    = (float*)alloc((size_t)4096 * 4096 * 4);
  float*          kf    = (float*)alloc((size_t)4096 * 1024 * 4);
  float*          vf    = (float*)alloc((size_t)4096 * 1024 * 4);
  unsigned short* q_bf  = (unsigned short*)alloc((size_t)4096 * 4096 * 2);
  unsigned short* k_bf  = (unsigned short*)alloc((size_t)1024 * 4096 * 2);
  unsigned short* vt_bf = (unsigned short*)alloc((size_t)1024 * 4096 * 2);

  // fp32 -> bf16 staging
  k_f32_to_bf16<<<65536, 256, 0, stream>>>(x,  x_bf,  16777216);
  k_f32_to_bf16<<<65536, 256, 0, stream>>>(wq, wq_bf, 16777216);
  k_f32_to_bf16<<<16384, 256, 0, stream>>>(wk, wk_bf,  4194304);
  k_f32_to_bf16<<<16384, 256, 0, stream>>>(wv, wv_bf,  4194304);
  k_f32_to_bf16<<<65536, 256, 0, stream>>>(wo, wo_bf, 16777216);

  // QKV projections (WMMA bf16, f32 accumulate)
  k_gemm_bf16<<<512, 256, 0, stream>>>(x_bf, wq_bf, qf, 4096, 4096, 4096);
  k_gemm_bf16<<<128, 256, 0, stream>>>(x_bf, wk_bf, kf, 4096, 1024, 4096);
  k_gemm_bf16<<<128, 256, 0, stream>>>(x_bf, wv_bf, vf, 4096, 1024, 4096);

  // RMSNorm + RoPE -> head-major bf16; V -> transposed bf16 (LDS tile transpose)
  k_normrope<<<16384, 256, 0, stream>>>(qf, qnw, cosb, sinb, q_bf, NH_);
  k_normrope<<< 4096, 256, 0, stream>>>(kf, knw, cosb, sinb, k_bf, NKV_);
  k_vtrans  <<< 4096, 256, 0, stream>>>(vf, vt_bf);

  // causal flash attention (attn output reuses x_bf region; x no longer needed)
  unsigned short* attn_bf = x_bf;
  k_flash<<<1024, 256, 0, stream>>>(q_bf, k_bf, vt_bf, attn_bf);

  // output projection -> d_out (fp32)
  k_gemm_bf16<<<512, 256, 0, stream>>>(attn_bf, wo_bf, out, 4096, 4096, 4096);
}
